// GNN_16535624089969
// MI455X (gfx1250) — matrix operand order repaired and checked by reading, not compiled, not run on hardware
//
#include <hip/hip_runtime.h>

// ---------------------------------------------------------------------------
// 2-layer GCN for MI455X (gfx1250, wave32, WMMA).
//   h1s  = (x @ W1) * norm_src                       (bf16 WMMA, f32 acc)
//   agg1 = segment_sum(h1s[src] -> dst)              (f32 atomics, L2-resident)
//   y    = relu(agg1*norm_dst + b1) * norm_src       (bf16)
//   h2s  = y @ W2                                    (bf16 WMMA, f32 acc)
//   out  = segment_sum(h2s[src] -> dst)*norm_dst + b2
// ---------------------------------------------------------------------------

typedef __attribute__((ext_vector_type(16))) __bf16 v16bf;
typedef __attribute__((ext_vector_type(8)))  float  v8f;
typedef unsigned short ushort_t;

union FragBF {
    uint4 q[2];   // 32 bytes = 16 bf16
    v16bf v;
};

__device__ __forceinline__ ushort_t f2bf(float f) {
    unsigned int u = __float_as_uint(f);
    unsigned int r = u + 0x7FFFu + ((u >> 16) & 1u);   // round-to-nearest-even
    return (ushort_t)(r >> 16);
}

// ------------------------- degrees & norms ---------------------------------
__global__ void gcn_degree_kernel(const int* __restrict__ src,
                                  const int* __restrict__ dst,
                                  float* __restrict__ dsrc,
                                  float* __restrict__ ddst, int E) {
    int e = blockIdx.x * 256 + threadIdx.x;
    if (e >= E) return;
    atomicAdd(dsrc + src[e], 1.0f);
    atomicAdd(ddst + dst[e], 1.0f);
}

__global__ void gcn_norm_kernel(float* __restrict__ ds,
                                float* __restrict__ dd, int n) {
    int i = blockIdx.x * 256 + threadIdx.x;
    if (i >= n) return;
    ds[i] = rsqrtf(fmaxf(ds[i], 1.0f));
    dd[i] = rsqrtf(fmaxf(dd[i], 1.0f));
}

// ------------------------- f32 -> bf16 convert -----------------------------
__global__ void gcn_cvt_bf16(const float* __restrict__ x,
                             ushort_t* __restrict__ xb, int n) {
    int i = blockIdx.x * 256 + threadIdx.x;
    if (i < n) xb[i] = f2bf(x[i]);
}

// ------------------------- pack W into B-fragment layout -------------------
// Packed layout: Wp[((nt*KC + kc)*32 + lane)*16 + e]
// B (K x 16) 16-bit fragment: K = kc*32 + hi*8 + {0..7, 16..23}[e]
// (hi = lane >> 4), column n = nt*16 + (lane & 15).
template <int K, int N>
__global__ void gcn_packB(const float* __restrict__ W,
                          ushort_t* __restrict__ Wp) {
    constexpr int KC = K / 32;
    constexpr int NT = N / 16;
    int idx = blockIdx.x * 256 + threadIdx.x;
    if (idx >= NT * KC * 32 * 16) return;
    int e    = idx & 15;
    int lane = (idx >> 4) & 31;
    int rest = idx >> 9;
    int kc   = rest % KC;
    int nt   = rest / KC;
    int k = kc * 32 + ((e >> 3) << 4) + ((lane >> 4) << 3) + (e & 7);
    int n = nt * 16 + (lane & 15);
    Wp[idx] = f2bf(W[k * N + n]);
}

// ------------------------- WMMA GEMM ---------------------------------------
// One wave per 16-row tile; A fragment loaded once per k-chunk and reused
// across all N/16 column tiles (8 f32 accumulators each).
// A (16x32 bf16) per-lane: row = lane&15, K = hi*8 + {0..7, 16..23}  -> two
// contiguous 16B loads from the row-major bf16 activation matrix.
template <int K, int N, bool SCALE>
__global__ void __launch_bounds__(128)
gcn_gemm_bf16(const ushort_t* __restrict__ A,
              const ushort_t* __restrict__ Bp,
              const float* __restrict__ rowscale,
              float* __restrict__ C, int Mtiles) {
    constexpr int KC = K / 32;
    constexpr int NT = N / 16;
    int wave = threadIdx.x >> 5;
    int lane = threadIdx.x & 31;
    int tile = blockIdx.x * 4 + wave;
    if (tile >= Mtiles) return;
    int m0 = tile * 16;
    int r  = lane & 15;
    int hi = lane >> 4;

    v8f acc[NT] = {};

    const ushort_t* arow = A + (size_t)(m0 + r) * K + hi * 8;
#pragma unroll
    for (int kc = 0; kc < KC; ++kc) {
        FragBF a;
        const uint4* ap = (const uint4*)(arow + kc * 32);
        a.q[0] = ap[0];     // K = kc*32 + hi*8 + 0..7
        a.q[1] = ap[2];     // K = kc*32 + hi*8 + 16..23
#pragma unroll
        for (int t = 0; t < NT; ++t) {
            FragBF b;
            const uint4* bp =
                (const uint4*)(Bp + ((size_t)(t * KC + kc) * 32 + lane) * 16);
            b.q[0] = bp[0];
            b.q[1] = bp[1];
            acc[t] = __builtin_amdgcn_wmma_f32_16x16x32_bf16(
                false, a.v, false, b.v, (short)0, acc[t], false, false);
        }
    }

    // C/D layout: VGPR i -> row m0 + hi*8 + i, col = lane&15
#pragma unroll
    for (int i = 0; i < 8; ++i) {
        int row = m0 + hi * 8 + i;
        float s = SCALE ? rowscale[row] : 1.0f;
#pragma unroll
        for (int t = 0; t < NT; ++t) {
            C[(size_t)row * N + t * 16 + r] = acc[t][i] * s;
        }
    }
}

// ------------------------- edge SpMM (atomic scatter) ----------------------
template <int F>
__global__ void gcn_spmm(const float* __restrict__ H,
                         const int* __restrict__ src,
                         const int* __restrict__ dst,
                         float* __restrict__ AGG, int E) {
    unsigned tid = blockIdx.x * 256u + threadIdx.x;
    unsigned e = tid / F;          // F is a power of two -> shift
    unsigned f = tid & (F - 1);
    if (e >= (unsigned)E) return;
    int s = src[e];
    int d = dst[e];
    atomicAdd(AGG + (size_t)d * F + f, H[(size_t)s * F + f]);
}

// ------------------------- epilogues ---------------------------------------
// y = relu(agg*norm_dst + b1) * norm_src, as bf16 (pre-scaled layer-2 input)
__global__ void gcn_finalize1(const float* __restrict__ agg,
                              const float* __restrict__ b1,
                              const float* __restrict__ nd,
                              const float* __restrict__ ns,
                              ushort_t* __restrict__ y, int total) {
    unsigned tid = blockIdx.x * 256u + threadIdx.x;
    if (tid >= (unsigned)total) return;
    unsigned n = tid >> 7;        // HID = 128
    unsigned f = tid & 127u;
    float v = agg[tid] * nd[n] + b1[f];
    v = fmaxf(v, 0.0f) * ns[n];
    y[tid] = f2bf(v);
}

// out = out*norm_dst + b2   (in place; out held the layer-2 aggregate)
__global__ void gcn_finalize2(float* __restrict__ out,
                              const float* __restrict__ b2,
                              const float* __restrict__ nd, int total) {
    unsigned tid = blockIdx.x * 256u + threadIdx.x;
    if (tid >= (unsigned)total) return;
    unsigned n = tid >> 6;        // OUT = 64
    unsigned f = tid & 63u;
    out[tid] = out[tid] * nd[n] + b2[f];
}

// ---------------------------------------------------------------------------
extern "C" void kernel_launch(void* const* d_in, const int* in_sizes, int n_in,
                              void* d_out, int out_size, void* d_ws,
                              size_t ws_size, hipStream_t stream) {
    const float* x  = (const float*)d_in[0];
    const float* W1 = (const float*)d_in[1];
    const float* b1 = (const float*)d_in[2];
    const float* W2 = (const float*)d_in[3];
    const float* b2 = (const float*)d_in[4];
    const int* src  = (const int*)d_in[5];
    const int* dst  = (const int*)d_in[6];

    constexpr int IN = 256, HID = 128, OUT = 64;
    const int Nn = in_sizes[0] / IN;   // 50000
    const int E  = in_sizes[5];        // 800000

    // ---- workspace layout (256B aligned) ----
    char* ws = (char*)d_ws;
    size_t off = 0;
    auto alloc = [&](size_t bytes) -> void* {
        void* p = ws + off;
        off += (bytes + 255) & ~(size_t)255;
        return p;
    };
    float*    norm_src = (float*)alloc((size_t)Nn * 4);
    float*    norm_dst = (float*)alloc((size_t)Nn * 4);
    ushort_t* W1p      = (ushort_t*)alloc((size_t)IN * HID * 2);
    ushort_t* W2p      = (ushort_t*)alloc((size_t)HID * OUT * 2);
    ushort_t* xb       = (ushort_t*)alloc((size_t)Nn * IN * 2);   // 25.6 MB
    float*    h1s      = (float*)alloc((size_t)Nn * HID * 4);     // 25.6 MB
    float*    agg1     = (float*)alloc((size_t)Nn * HID * 4);     // 25.6 MB
    // aliases (lifetimes do not overlap):
    ushort_t* ybf = xb;              // layer-2 bf16 input  (Nn*HID*2 <= Nn*IN*2)
    float*    h2s = h1s;             // layer-2 GEMM output (Nn*OUT*4 <= Nn*HID*4)

    // ---- zero what we accumulate into ----
    hipMemsetAsync(norm_src, 0, (size_t)Nn * 4, stream);
    hipMemsetAsync(norm_dst, 0, (size_t)Nn * 4, stream);
    hipMemsetAsync(agg1, 0, (size_t)Nn * HID * 4, stream);
    hipMemsetAsync(d_out, 0, (size_t)out_size * 4, stream);

    // ---- degrees -> norms ----
    gcn_degree_kernel<<<(E + 255) / 256, 256, 0, stream>>>(src, dst, norm_src,
                                                           norm_dst, E);
    gcn_norm_kernel<<<(Nn + 255) / 256, 256, 0, stream>>>(norm_src, norm_dst,
                                                          Nn);

    // ---- convert activations, pack weights ----
    {
        int n = Nn * IN;
        gcn_cvt_bf16<<<(n + 255) / 256, 256, 0, stream>>>(x, xb, n);
    }
    gcn_packB<IN, HID><<<(IN * HID + 255) / 256, 256, 0, stream>>>(W1, W1p);
    gcn_packB<HID, OUT><<<(HID * OUT + 255) / 256, 256, 0, stream>>>(W2, W2p);

    const int Mtiles = (Nn + 15) / 16;            // 3125
    const int gemmBlocks = (Mtiles + 3) / 4;

    // ---- layer 1 ----
    gcn_gemm_bf16<IN, HID, true><<<gemmBlocks, 128, 0, stream>>>(
        xb, W1p, norm_src, h1s, Mtiles);
    {
        size_t total = (size_t)E * HID;
        gcn_spmm<HID><<<(unsigned)((total + 255) / 256), 256, 0, stream>>>(
            h1s, src, dst, agg1, E);
    }
    {
        int total = Nn * HID;
        gcn_finalize1<<<(total + 255) / 256, 256, 0, stream>>>(
            agg1, b1, norm_dst, norm_src, ybf, total);
    }

    // ---- layer 2 ----
    gcn_gemm_bf16<HID, OUT, false><<<gemmBlocks, 128, 0, stream>>>(
        ybf, W2p, nullptr, h2s, Mtiles);
    {
        size_t total = (size_t)E * OUT;
        gcn_spmm<OUT><<<(unsigned)((total + 255) / 256), 256, 0, stream>>>(
            h2s, src, dst, (float*)d_out, E);
    }
    {
        int total = Nn * OUT;
        gcn_finalize2<<<(total + 255) / 256, 256, 0, stream>>>(
            (float*)d_out, b2, norm_dst, total);
    }
}
